// AttentionBlock_21440476742190
// MI455X (gfx1250) — matrix operand order repaired and checked by reading, not compiled, not run on hardware
//
#include <hip/hip_runtime.h>
#include <hip/hip_bf16.h>
#include <cstdint>

typedef unsigned short u16;
typedef __attribute__((ext_vector_type(16))) __bf16 v16bf;
typedef __attribute__((ext_vector_type(8)))  float  v8f;

#define NB   4
#define NC   256
#define NSP  32768     // 32*32*32 spatial
#define NG   4
#define CPG  64        // channels per group
#define NH   8
#define HD   32
#define QKVO 768

// ---------- bf16 helpers (hardware cvt via __bf16) ----------
static __device__ __forceinline__ u16 f2bf(float f) {
  return __builtin_bit_cast(u16, (__bf16)f);
}
static __device__ __forceinline__ float bf2f(u16 h) {
  union { unsigned u; float f; } v; v.u = ((unsigned)h) << 16;
  return v.f;
}
struct BFPair { uint4 lo, hi; };
static __device__ __forceinline__ v16bf make_frag(uint4 lo, uint4 hi) {
  BFPair t; t.lo = lo; t.hi = hi;
  return __builtin_bit_cast(v16bf, t);
}

// ---------- K1: GroupNorm partial sums ----------
__global__ __launch_bounds__(256) void gn_partial(const float* __restrict__ x,
                                                  float* __restrict__ stats) {
  const int bg    = blockIdx.x >> 6;   // (b*NG+g)
  const int chunk = blockIdx.x & 63;
  const int tid   = threadIdx.x;
  const float4* p = (const float4*)(x + (size_t)bg * ((size_t)CPG * NSP)) +
                    (size_t)chunk * 8192;
  float s = 0.f, ss = 0.f;
  for (int i = tid; i < 8192; i += 256) {
    float4 v = p[i];
    s  += v.x + v.y + v.z + v.w;
    ss += v.x*v.x + v.y*v.y + v.z*v.z + v.w*v.w;
  }
  __shared__ float sh[512];
  sh[tid] = s; sh[256 + tid] = ss;
  __syncthreads();
  for (int st = 128; st > 0; st >>= 1) {
    if (tid < st) { sh[tid] += sh[tid + st]; sh[256 + tid] += sh[256 + tid + st]; }
    __syncthreads();
  }
  if (tid == 0) {
    atomicAdd(&stats[bg * 2 + 0], sh[0]);
    atomicAdd(&stats[bg * 2 + 1], sh[256]);
  }
}

// ---------- K2: finalize mean / rstd ----------
__global__ void gn_finalize(const float* __restrict__ stats, float* __restrict__ mr) {
  const int t = threadIdx.x;
  if (t < NB * NG) {
    const float cnt  = (float)((size_t)CPG * NSP);
    const float mean = stats[t * 2] / cnt;
    const float var  = stats[t * 2 + 1] / cnt - mean * mean;
    mr[t * 2 + 0] = mean;
    mr[t * 2 + 1] = rsqrtf(var + 1e-5f);
  }
}

// ---------- util: fp32 -> bf16 convert ----------
__global__ void cvt_bf16(const float* __restrict__ in, u16* __restrict__ out, int n) {
  const int i = blockIdx.x * 256 + threadIdx.x;
  if (i < n) out[i] = f2bf(in[i]);
}

// ---------- K3: fused GroupNorm-apply + QKV GEMM (bf16 WMMA, double-buffered) ----------
// grid: (NSP/64, QKVO/128, NB), block 256 (8 waves). Tile: M=128(o) x N=64(n), K=256(c)
__global__ __launch_bounds__(256) void qkv_gemm(
    const float* __restrict__ x, const float* __restrict__ mr,
    const float* __restrict__ gnw, const float* __restrict__ gnb,
    const u16* __restrict__ wq,
    u16* __restrict__ qo, u16* __restrict__ ko, u16* __restrict__ vo) {
  // ping-pong: buf i at smem + i*6144; A = +0 (128*32), B = +4096 (64*32)
  __shared__ __align__(16) u16 smem[2 * 6144];     // 24 KB
  const int tid    = threadIdx.x;
  const int n0     = blockIdx.x * 64;
  const int obase  = blockIdx.y * 128;
  const int b      = blockIdx.z;
  const int wave   = tid >> 5, lane = tid & 31;
  const int lanelo = lane & 15, lanehi = lane >> 4;
  const int wm     = wave >> 1, wn = wave & 1;

  v8f acc[2][2] = {};
  const int arow = tid >> 1, ahalf = tid & 1;        // A stage: 128 rows x 2 halves
  const int bcl  = tid >> 3, bnc   = (tid & 7) * 8;  // B stage: 32 c-rows x 8-float chunks

  uint4 ra0, ra1; float4 rb0, rb1;
  auto gload = [&](int kc) {
    const int cb = kc * 32;
    const u16* srcA = wq + ((size_t)(obase + arow) * NC + cb + ahalf * 16);
    ra0 = *(const uint4*)srcA;
    ra1 = *(const uint4*)(srcA + 8);
    const float* srcB = x + ((size_t)(b * NC + cb + bcl) * NSP + n0 + bnc);
    rb0 = *(const float4*)srcB;
    rb1 = *(const float4*)(srcB + 4);
    if (kc < 7) {
      __builtin_prefetch(srcA + NC * 0 + 32, 0, 3);   // next K-chunk of weights
      __builtin_prefetch(srcB + (size_t)32 * NSP, 0, 3); // next K-chunk of x
    }
  };
  auto sstore = [&](int kc, int bufi) {
    u16* A = smem + bufi * 6144;
    u16* B = A + 4096;
    *(uint4*)(A + arow * 32 + ahalf * 16)     = ra0;
    *(uint4*)(A + arow * 32 + ahalf * 16 + 8) = ra1;
    const int c = kc * 32 + bcl;
    const int g = c >> 6;
    const float mean = mr[(b * NG + g) * 2];
    const float rstd = mr[(b * NG + g) * 2 + 1];
    const float sc = rstd * gnw[c];
    const float sh = gnb[c] - mean * sc;
    u16* dst = B + bnc * 32 + bcl;    // transposed [n][c]
    dst[0 * 32] = f2bf(rb0.x * sc + sh); dst[1 * 32] = f2bf(rb0.y * sc + sh);
    dst[2 * 32] = f2bf(rb0.z * sc + sh); dst[3 * 32] = f2bf(rb0.w * sc + sh);
    dst[4 * 32] = f2bf(rb1.x * sc + sh); dst[5 * 32] = f2bf(rb1.y * sc + sh);
    dst[6 * 32] = f2bf(rb1.z * sc + sh); dst[7 * 32] = f2bf(rb1.w * sc + sh);
  };

  gload(0);
  sstore(0, 0);
  __syncthreads();
  for (int kc = 0; kc < 8; ++kc) {
    const int cur = kc & 1;
    if (kc < 7) gload(kc + 1);
    const u16* A = smem + cur * 6144;
    const u16* B = A + 4096;
    v16bf a[2], bf[2];
#pragma unroll
    for (int i = 0; i < 2; ++i) {
      const u16* ap = A + (wm * 32 + i * 16 + lanelo) * 32;
      a[i] = make_frag(*(const uint4*)(ap + 16 * lanehi),
                       *(const uint4*)(ap + 16 * lanehi + 8));
    }
#pragma unroll
    for (int j = 0; j < 2; ++j) {
      const u16* bp = B + (wn * 32 + j * 16 + lanelo) * 32 + 16 * lanehi;
      bf[j] = make_frag(*(const uint4*)bp, *(const uint4*)(bp + 8));
    }
#pragma unroll
    for (int i = 0; i < 2; ++i)
#pragma unroll
      for (int j = 0; j < 2; ++j)
        acc[i][j] = __builtin_amdgcn_wmma_f32_16x16x32_bf16(
            false, a[i], false, bf[j], (short)0, acc[i][j], false, false);
    if (kc < 7) sstore(kc + 1, cur ^ 1);
    __syncthreads();
  }
  // epilogue: repack D through LDS so global writes are 64B-coalesced b128s
  u16* rep = smem;                                  // 128*64 u16 = 16 KB (fits)
#pragma unroll
  for (int i = 0; i < 2; ++i)
#pragma unroll
    for (int j = 0; j < 2; ++j) {
      const int nl = wn * 32 + j * 16 + lanelo;
#pragma unroll
      for (int r = 0; r < 8; ++r) {
        const int ol = wm * 32 + i * 16 + r + 8 * lanehi;
        rep[ol * 64 + nl] = f2bf(acc[i][j][r]);
      }
    }
  __syncthreads();
  u16* dstArr = (blockIdx.y < 2) ? qo : ((blockIdx.y < 4) ? ko : vo);
  const int row  = tid >> 1;                 // 0..127 (o_local)
  const int half = (tid & 1) * 32;           // 0 or 32 (n_local)
  const int cp   = (obase + row) & 255;
  const uint4* sv = (const uint4*)(rep + row * 64 + half);
  uint4* gv = (uint4*)(dstArr + ((size_t)(b * NC + cp)) * NSP + n0 + half);
#pragma unroll
  for (int q = 0; q < 4; ++q) gv[q] = sv[q];
}

// ---------- K4: softmax over N per (b,h,d) row, register-cached single pass ----------
__global__ __launch_bounds__(256) void softmax_rows(u16* __restrict__ k) {
  const size_t base = (size_t)blockIdx.x * NSP;
  const int tid = threadIdx.x;
  __shared__ float sh[256];
  uint4* kv = (uint4*)(k + base);                 // 4096 uint4 (8 bf16 each)
  uint4 vals[16];                                 // 128 bf16 per thread, read once
#pragma unroll
  for (int i = 0; i < 16; ++i) vals[i] = kv[tid + i * 256];

  float m = -3.4e38f;
#pragma unroll
  for (int i = 0; i < 16; ++i) {
    const unsigned w[4] = { vals[i].x, vals[i].y, vals[i].z, vals[i].w };
#pragma unroll
    for (int q = 0; q < 4; ++q) {
      m = fmaxf(m, bf2f((u16)(w[q] & 0xffff)));
      m = fmaxf(m, bf2f((u16)(w[q] >> 16)));
    }
  }
  sh[tid] = m; __syncthreads();
  for (int st = 128; st; st >>= 1) {
    if (tid < st) sh[tid] = fmaxf(sh[tid], sh[tid + st]);
    __syncthreads();
  }
  const float rowmax = sh[0]; __syncthreads();
  float s = 0.f;
#pragma unroll
  for (int i = 0; i < 16; ++i) {
    const unsigned w[4] = { vals[i].x, vals[i].y, vals[i].z, vals[i].w };
#pragma unroll
    for (int q = 0; q < 4; ++q) {
      s += __expf(bf2f((u16)(w[q] & 0xffff)) - rowmax);
      s += __expf(bf2f((u16)(w[q] >> 16))   - rowmax);
    }
  }
  sh[tid] = s; __syncthreads();
  for (int st = 128; st; st >>= 1) {
    if (tid < st) sh[tid] += sh[tid + st];
    __syncthreads();
  }
  const float inv = 1.f / sh[0];
#pragma unroll
  for (int i = 0; i < 16; ++i) {
    unsigned w[4] = { vals[i].x, vals[i].y, vals[i].z, vals[i].w };
#pragma unroll
    for (int q = 0; q < 4; ++q) {
      const u16 lo = f2bf(__expf(bf2f((u16)(w[q] & 0xffff)) - rowmax) * inv);
      const u16 hi = f2bf(__expf(bf2f((u16)(w[q] >> 16))   - rowmax) * inv);
      w[q] = (unsigned)lo | ((unsigned)hi << 16);
    }
    kv[tid + i * 256] = make_uint4(w[0], w[1], w[2], w[3]);
  }
}

// ---------- K5: sim[b,h] = k' (32 x N) * v^T (N x 32), WMMA over N ----------
__global__ __launch_bounds__(256) void sim_kernel(const u16* __restrict__ k,
                                                  const u16* __restrict__ v,
                                                  float* __restrict__ sim) {
  const int bh = blockIdx.x;                       // b*NH+h; channel base = bh*32
  const u16* kb = k + (size_t)bh * HD * NSP;
  const u16* vb = v + (size_t)bh * HD * NSP;
  const int tid = threadIdx.x;
  const int wave = tid >> 5, lane = tid & 31;
  const int lanelo = lane & 15, lanehi = lane >> 4;
  v8f acc[2][2] = {};
  for (int t = wave; t < NSP / 32; t += 8) {
    const int n0 = t * 32;
    v16bf a[2], bf[2];
#pragma unroll
    for (int i = 0; i < 2; ++i) {
      const u16* ap = kb + (size_t)(i * 16 + lanelo) * NSP + n0;
      a[i] = make_frag(*(const uint4*)(ap + 16 * lanehi),
                       *(const uint4*)(ap + 16 * lanehi + 8));
    }
#pragma unroll
    for (int j = 0; j < 2; ++j) {
      const u16* bp = vb + (size_t)(j * 16 + lanelo) * NSP + n0 + 16 * lanehi;
      bf[j] = make_frag(*(const uint4*)bp, *(const uint4*)(bp + 8));
    }
#pragma unroll
    for (int i = 0; i < 2; ++i)
#pragma unroll
      for (int j = 0; j < 2; ++j)
        acc[i][j] = __builtin_amdgcn_wmma_f32_16x16x32_bf16(
            false, a[i], false, bf[j], (short)0, acc[i][j], false, false);
  }
  __shared__ float red[8 * 32 * 32];               // 32 KB
#pragma unroll
  for (int i = 0; i < 2; ++i)
#pragma unroll
    for (int j = 0; j < 2; ++j)
#pragma unroll
      for (int r = 0; r < 8; ++r) {
        const int mrow = i * 16 + r + 8 * lanehi;
        const int ncol = j * 16 + lanelo;
        red[wave * 1024 + mrow * 32 + ncol] = acc[i][j][r];
      }
  __syncthreads();
  for (int idx = tid; idx < 1024; idx += 256) {
    float s = 0.f;
#pragma unroll
    for (int w = 0; w < 8; ++w) s += red[w * 1024 + idx];
    sim[(size_t)bh * 1024 + idx] = s;              // sim[b,h][d][e]
  }
}

// ---------- K6: fold sim into out_weight: Wf[b][o][(h,d)] = sum_e W[o][(h,e)] sim[b,h,d,e]
__global__ void wfused_kernel(const float* __restrict__ ow,
                              const float* __restrict__ sim,
                              u16* __restrict__ wf) {
  const int idx = blockIdx.x * 256 + threadIdx.x;  // NB*256*256
  const int b  = idx >> 16;
  const int o  = (idx >> 8) & 255;
  const int cp = idx & 255;
  const int h = cp >> 5, d = cp & 31;
  const float* wrow = ow + o * NC + h * 32;
  const float* srow = sim + (((size_t)(b * NH + h)) * 32 + d) * 32;
  float s = 0.f;
#pragma unroll
  for (int e = 0; e < 32; ++e) s += wrow[e] * srow[e];
  wf[idx] = f2bf(s);
}

// ---------- K7: out = Wfused_b (256x256) * q_b (256xN) + bias + x (double-buffered) ----------
__global__ __launch_bounds__(256) void out_gemm(
    const u16* __restrict__ wf, const u16* __restrict__ q,
    const float* __restrict__ bias, const float* __restrict__ x,
    float* __restrict__ out) {
  __shared__ __align__(16) u16 smem[2 * 6144];     // 24 KB
  const int tid    = threadIdx.x;
  const int n0     = blockIdx.x * 64;
  const int obase  = blockIdx.y * 128;
  const int b      = blockIdx.z;
  const int wave   = tid >> 5, lane = tid & 31;
  const int lanelo = lane & 15, lanehi = lane >> 4;
  const int wm     = wave >> 1, wn = wave & 1;

  v8f acc[2][2] = {};
  const int arow = tid >> 1, ahalf = tid & 1;
  const int bcl  = tid >> 3, bnc   = (tid & 7) * 8;
  const u16* wfb = wf + (size_t)b * NC * NC;

  uint4 ra0, ra1, rbq;
  auto gload = [&](int kc) {
    const int cb = kc * 32;
    const u16* srcA = wfb + ((size_t)(obase + arow) * NC + cb + ahalf * 16);
    ra0 = *(const uint4*)srcA;
    ra1 = *(const uint4*)(srcA + 8);
    const u16* srcB = q + ((size_t)(b * NC + cb + bcl)) * NSP + n0 + bnc;
    rbq = *(const uint4*)srcB;
    if (kc < 7) {
      __builtin_prefetch(srcA + 32, 0, 3);
      __builtin_prefetch(srcB + (size_t)32 * NSP, 0, 3);
    }
  };
  auto sstore = [&](int bufi) {
    u16* A = smem + bufi * 6144;
    u16* B = A + 4096;
    *(uint4*)(A + arow * 32 + ahalf * 16)     = ra0;
    *(uint4*)(A + arow * 32 + ahalf * 16 + 8) = ra1;
    u16* dst = B + bnc * 32 + bcl;   // transposed [n][c]
    dst[0 * 32] = (u16)(rbq.x & 0xffff); dst[1 * 32] = (u16)(rbq.x >> 16);
    dst[2 * 32] = (u16)(rbq.y & 0xffff); dst[3 * 32] = (u16)(rbq.y >> 16);
    dst[4 * 32] = (u16)(rbq.z & 0xffff); dst[5 * 32] = (u16)(rbq.z >> 16);
    dst[6 * 32] = (u16)(rbq.w & 0xffff); dst[7 * 32] = (u16)(rbq.w >> 16);
  };

  gload(0);
  sstore(0);
  __syncthreads();
  for (int kc = 0; kc < 8; ++kc) {
    const int cur = kc & 1;
    if (kc < 7) gload(kc + 1);
    const u16* A = smem + cur * 6144;
    const u16* B = A + 4096;
    v16bf a[2], bf[2];
#pragma unroll
    for (int i = 0; i < 2; ++i) {
      const u16* ap = A + (wm * 32 + i * 16 + lanelo) * 32;
      a[i] = make_frag(*(const uint4*)(ap + 16 * lanehi),
                       *(const uint4*)(ap + 16 * lanehi + 8));
    }
#pragma unroll
    for (int j = 0; j < 2; ++j) {
      const u16* bp = B + (wn * 32 + j * 16 + lanelo) * 32 + 16 * lanehi;
      bf[j] = make_frag(*(const uint4*)bp, *(const uint4*)(bp + 8));
    }
#pragma unroll
    for (int i = 0; i < 2; ++i)
#pragma unroll
      for (int j = 0; j < 2; ++j)
        acc[i][j] = __builtin_amdgcn_wmma_f32_16x16x32_bf16(
            false, a[i], false, bf[j], (short)0, acc[i][j], false, false);
    if (kc < 7) sstore(cur ^ 1);
    __syncthreads();
  }
  // epilogue: + bias + residual, fp32 out (16-lane rows -> 64B segments)
#pragma unroll
  for (int i = 0; i < 2; ++i)
#pragma unroll
    for (int j = 0; j < 2; ++j) {
      const int n = n0 + wn * 32 + j * 16 + lanelo;
#pragma unroll
      for (int r = 0; r < 8; ++r) {
        const int o = obase + wm * 32 + i * 16 + r + 8 * lanehi;
        const size_t addr = ((size_t)(b * NC + o)) * NSP + n;
        out[addr] = acc[i][j][r] + bias[o] + x[addr];
      }
    }
}

// ---------- launcher ----------
extern "C" void kernel_launch(void* const* d_in, const int* in_sizes, int n_in,
                              void* d_out, int out_size, void* d_ws, size_t ws_size,
                              hipStream_t stream) {
  const float* x    = (const float*)d_in[0];
  const float* gnw  = (const float*)d_in[1];
  const float* gnb  = (const float*)d_in[2];
  const float* wqkv = (const float*)d_in[3];
  const float* ow   = (const float*)d_in[4];
  const float* ob   = (const float*)d_in[5];
  float* out = (float*)d_out;

  char* ws = (char*)d_ws;
  const size_t qkv_elems = (size_t)NB * NC * NSP;          // per q/k/v, bf16
  float* stats = (float*)(ws + 0);                          // 128 B
  float* mr    = (float*)(ws + 256);                        // 128 B
  u16*   wq_bf = (u16*)(ws + 512);                          // 768*256*2
  u16*   q_ws  = (u16*)(ws + 512 + (size_t)QKVO * NC * 2);
  u16*   k_ws  = q_ws + qkv_elems;
  u16*   v_ws  = k_ws + qkv_elems;
  float* sim   = (float*)((char*)(v_ws + qkv_elems));       // NB*NH*32*32 f32
  u16*   wfu   = (u16*)((char*)sim + (size_t)NB * NH * 32 * 32 * 4);

  hipMemsetAsync(stats, 0, 32 * sizeof(float), stream);
  gn_partial   <<<NB * NG * 64, 256, 0, stream>>>(x, stats);
  gn_finalize  <<<1, 64, 0, stream>>>(stats, mr);
  cvt_bf16     <<<(QKVO * NC + 255) / 256, 256, 0, stream>>>(wqkv, wq_bf, QKVO * NC);
  qkv_gemm     <<<dim3(NSP / 64, QKVO / 128, NB), 256, 0, stream>>>(
                   x, mr, gnw, gnb, wq_bf, q_ws, k_ws, v_ws);
  softmax_rows <<<NB * NC, 256, 0, stream>>>(k_ws);
  sim_kernel   <<<NB * NH, 256, 0, stream>>>(k_ws, v_ws, sim);
  wfused_kernel<<<NB * NC * NC / 256, 256, 0, stream>>>(ow, sim, wfu);
  out_gemm     <<<dim3(NSP / 64, NC / 128, NB), 256, 0, stream>>>(wfu, q_ws, ob, x, out);
}
